// SelfAttentionBlock_19688130085286
// MI455X (gfx1250) — matrix operand order repaired and checked by reading, not compiled, not run on hardware
//
#include <hip/hip_runtime.h>

typedef __attribute__((ext_vector_type(16))) __bf16          v16bf;
typedef __attribute__((ext_vector_type(8)))  float           v8f;
typedef __attribute__((ext_vector_type(16))) unsigned short  v16u;
typedef __attribute__((ext_vector_type(8)))  unsigned short  v8u;

#define S_TOT 6144
#define C_TOT 128

// hardware RNE f32 -> bf16 (native v_cvt on gfx1250)
__device__ __forceinline__ unsigned short bf16_of_f32(float f) {
  __bf16 h = (__bf16)f;
  return __builtin_bit_cast(unsigned short, h);
}

// cross-lane XOR exchange within the wave via LDS-permute hardware (no LDS mem)
#define XORF(v, m) __int_as_float(__builtin_amdgcn_ds_swizzle(__float_as_int(v), (((m) << 10) | 0x1f)))

// ---------------------------------------------------------------------------
// Kernel 0: one-shot f32 -> bf16 conversion of both weight matrices so the
// GEMM kernels do pure contiguous loads + WMMA (no per-tile conversion VALU).
// ---------------------------------------------------------------------------
__global__ void k_wprep(const float* __restrict__ w0, int n0,
                        const float* __restrict__ w1, int n1,
                        unsigned short* __restrict__ o0,
                        unsigned short* __restrict__ o1) {
  int i = blockIdx.x * 256 + threadIdx.x;
  if (i < n0) o0[i] = bf16_of_f32(w0[i]);
  if (i < n1) o1[i] = bf16_of_f32(w1[i]);
}

// ---------------------------------------------------------------------------
// Kernel 1: GroupNorm (8 groups x 16ch x 6144) -> xnT[s][c] bf16 (transposed
// so QKV-GEMM B fragments are contiguous 32B loads along the K=c axis)
// ---------------------------------------------------------------------------
__global__ void k_groupnorm(const float* __restrict__ x,
                            const float* __restrict__ gn_w,
                            const float* __restrict__ gn_b,
                            unsigned short* __restrict__ xnT) {
  __shared__ float s_sum[256], s_sq[256];
  const int g = blockIdx.x, tid = threadIdx.x;
  const int NE = 16 * S_TOT;             // 98304 elements per group
  const float* xg = x + g * NE;
  float sum = 0.f, sq = 0.f;
  for (int i = tid; i < NE; i += 256) { float v = xg[i]; sum += v; sq += v * v; }
  s_sum[tid] = sum; s_sq[tid] = sq; __syncthreads();
  for (int off = 128; off > 0; off >>= 1) {
    if (tid < off) { s_sum[tid] += s_sum[tid + off]; s_sq[tid] += s_sq[tid + off]; }
    __syncthreads();
  }
  const float mean = s_sum[0] * (1.0f / 98304.0f);
  const float var  = s_sq[0] * (1.0f / 98304.0f) - mean * mean;
  const float rstd = rsqrtf(var + 1e-5f);
  for (int i = tid; i < NE; i += 256) {
    int cl = i / S_TOT, s = i - cl * S_TOT;
    int c = g * 16 + cl;
    float v = (xg[i] - mean) * rstd * gn_w[c] + gn_b[c];
    xnT[s * C_TOT + c] = bf16_of_f32(v);
  }
}

// build a 16-element A/B fragment from two contiguous 8x bf16 runs
__device__ __forceinline__ v16bf frag_from_runs(const unsigned short* p) {
  v8u lo = *(const v8u*)p;
  v8u hh = *(const v8u*)(p + 16);
  v16u t;
#pragma unroll
  for (int e = 0; e < 8; e++) { t[e] = lo[e]; t[8 + e] = hh[e]; }
  return __builtin_bit_cast(v16bf, t);
}

// ---------------------------------------------------------------------------
// Kernel 2: QKV GEMM  (384x128) @ (128x6144), WMMA bf16 -> f32.
// Writes Q[h][s][c]*scale, K[h][s][c], V transposed Vt[h][c][s] (all bf16).
// Tile indices are wave-uniform -> readfirstlane so q/k/v split is SALU.
// ---------------------------------------------------------------------------
__global__ void k_qkv(const unsigned short* __restrict__ Wbf,
                      const float* __restrict__ bias,
                      const unsigned short* __restrict__ xnT,
                      unsigned short* __restrict__ Qb, unsigned short* __restrict__ Kb,
                      unsigned short* __restrict__ Vt) {
  const int lane = threadIdx.x & 31;
  const int t  = __builtin_amdgcn_readfirstlane(blockIdx.x * 4 + (threadIdx.x >> 5));
  const int mt = t % 24, nt = t / 24;        // scalar
  const int li = lane & 15, hi = lane >> 4;
  const int row = mt * 16 + li;
  const int s   = nt * 16 + li;
  v8f acc = {};
#pragma unroll
  for (int ks = 0; ks < 4; ks++) {
    v16bf a = __builtin_bit_cast(v16bf, *(const v16u*)(Wbf + row * C_TOT + ks * 32 + hi * 16));
    v16bf b = __builtin_bit_cast(v16bf, *(const v16u*)(xnT + s * C_TOT + ks * 32 + hi * 16));
    acc = __builtin_amdgcn_wmma_f32_16x16x32_bf16(false, a, false, b, (short)0, acc, false, false);
  }
  const int which  = mt >> 3;                // scalar: 0=q 1=k 2=v
  const int baseoc = (mt * 16) & 127;
  const int h      = baseoc >> 5;            // scalar head
  const int c0     = (baseoc & 31) + 8 * hi; // per-lane head-dim base (8-aligned)
  const int obase  = mt * 16 + 8 * hi;
  if (which == 2) {
#pragma unroll
    for (int r = 0; r < 8; r++) {
      float v = acc[r] + bias[obase + r];
      Vt[(h * 32 + c0 + r) * S_TOT + s] = bf16_of_f32(v);   // transposed: scattered
    }
  } else {
    const float sc = (which == 0) ? 0.17677669529663687f : 1.0f;
    v8u st;
#pragma unroll
    for (int r = 0; r < 8; r++) st[r] = bf16_of_f32((acc[r] + bias[obase + r]) * sc);
    unsigned short* dst = (which == 0) ? Qb : Kb;
    *(v8u*)(dst + (h * S_TOT + s) * 32 + c0) = st;          // one 16B store
  }
}

// ---------------------------------------------------------------------------
// Kernel 3: flash attention. One wave per (head, 16-row tile), streams all
// 6144 columns in 32-col steps: 2 WMMA for scores + 2 WMMA for P@V per step.
// Online softmax in f32; P staged through LDS to convert D-layout->A-layout.
// ---------------------------------------------------------------------------
__global__ void k_attn(const unsigned short* __restrict__ Qb,
                       const unsigned short* __restrict__ Kb,
                       const unsigned short* __restrict__ Vt,
                       unsigned short* __restrict__ attnT) {
  __shared__ unsigned short pbuf[4][16 * 32];
  const int lane = threadIdx.x & 31, wave = threadIdx.x >> 5;
  const int rt = __builtin_amdgcn_readfirstlane(blockIdx.x * 4 + wave);  // 0..383
  const int h  = blockIdx.y;                  // 0..3 (scalar)
  const int li = lane & 15, hi = lane >> 4;
  unsigned short* pb = pbuf[wave];

  // Q A-fragment (held for whole row tile)
  v16bf qa = __builtin_bit_cast(v16bf, *(const v16u*)(Qb + (h * S_TOT + rt * 16 + li) * 32 + hi * 16));

  float m[8], l[8];
  v8f o0 = {}, o1 = {};
#pragma unroll
  for (int r = 0; r < 8; r++) { m[r] = -3.0e38f; l[r] = 0.0f; }
  const float L2E = 1.4426950408889634f;

  for (int j0 = 0; j0 < S_TOT; j0 += 32) {
    const unsigned short* kp0 = Kb + (h * S_TOT + j0 + li) * 32 + hi * 16;
    const unsigned short* vp0 = Vt + (h * 32 + li) * S_TOT + j0 + hi * 16;
    // speculative prefetch of next iteration's K/V tiles (global_prefetch_b8)
    __builtin_prefetch(kp0 + 32 * 32, 0, 1);
    __builtin_prefetch(vp0 + 32, 0, 1);

    v16bf b0 = __builtin_bit_cast(v16bf, *(const v16u*)kp0);
    v16bf b1 = __builtin_bit_cast(v16bf, *(const v16u*)(kp0 + 16 * 32));
    v8f z = {};
    v8f s0 = __builtin_amdgcn_wmma_f32_16x16x32_bf16(false, qa, false, b0, (short)0, z, false, false);
    v8f s1 = __builtin_amdgcn_wmma_f32_16x16x32_bf16(false, qa, false, b1, (short)0, z, false, false);
#pragma unroll
    for (int r = 0; r < 8; r++) {
      float mx = fmaxf(s0[r], s1[r]);
      mx = fmaxf(mx, XORF(mx, 1)); mx = fmaxf(mx, XORF(mx, 2));
      mx = fmaxf(mx, XORF(mx, 4)); mx = fmaxf(mx, XORF(mx, 8));
      float mn = fmaxf(m[r], mx);
      float al = exp2f((m[r] - mn) * L2E);
      m[r] = mn;
      l[r] *= al; o0[r] *= al; o1[r] *= al;
      float p0 = exp2f((s0[r] - mn) * L2E);
      float p1 = exp2f((s1[r] - mn) * L2E);
      l[r] += p0 + p1;
      int M = r + 8 * hi;
      pb[M * 32 + li]      = bf16_of_f32(p0);
      pb[M * 32 + 16 + li] = bf16_of_f32(p1);
    }
    __builtin_amdgcn_wave_barrier();       // stores -> loads ordering (in-order LDS)
    v16bf pa  = __builtin_bit_cast(v16bf, *(const v16u*)(pb + li * 32 + hi * 16));
    v16bf vb0 = __builtin_bit_cast(v16bf, *(const v16u*)vp0);
    v16bf vb1 = __builtin_bit_cast(v16bf, *(const v16u*)(vp0 + 16 * S_TOT));
    o0 = __builtin_amdgcn_wmma_f32_16x16x32_bf16(false, pa, false, vb0, (short)0, o0, false, false);
    o1 = __builtin_amdgcn_wmma_f32_16x16x32_bf16(false, pa, false, vb1, (short)0, o1, false, false);
    __builtin_amdgcn_wave_barrier();
  }
#pragma unroll
  for (int r = 0; r < 8; r++) {
    float ls = l[r];
    ls += XORF(ls, 1); ls += XORF(ls, 2); ls += XORF(ls, 4); ls += XORF(ls, 8);
    float inv = 1.0f / ls;
    int srow = rt * 16 + r + 8 * hi;
    attnT[srow * C_TOT + h * 32 + li]      = bf16_of_f32(o0[r] * inv);
    attnT[srow * C_TOT + h * 32 + 16 + li] = bf16_of_f32(o1[r] * inv);
  }
}

// ---------------------------------------------------------------------------
// Kernel 4: proj GEMM (128x128)@(128x6144) + bias + residual + mask -> f32 out
// ---------------------------------------------------------------------------
__global__ void k_proj(const unsigned short* __restrict__ Wbf,
                       const float* __restrict__ bias,
                       const unsigned short* __restrict__ attnT,
                       const float* __restrict__ x, const float* __restrict__ mask,
                       float* __restrict__ out) {
  const int lane = threadIdx.x & 31;
  const int t  = __builtin_amdgcn_readfirstlane(blockIdx.x * 4 + (threadIdx.x >> 5));
  const int mt = t % 8, nt = t / 8;          // scalar
  const int li = lane & 15, hi = lane >> 4;
  const int row = mt * 16 + li;
  const int s   = nt * 16 + li;
  v8f acc = {};
#pragma unroll
  for (int ks = 0; ks < 4; ks++) {
    v16bf a = __builtin_bit_cast(v16bf, *(const v16u*)(Wbf + row * C_TOT + ks * 32 + hi * 16));
    v16bf b = __builtin_bit_cast(v16bf, *(const v16u*)(attnT + s * C_TOT + ks * 32 + hi * 16));
    acc = __builtin_amdgcn_wmma_f32_16x16x32_bf16(false, a, false, b, (short)0, acc, false, false);
  }
  const float mk = mask[s];
  const int obase = mt * 16 + 8 * hi;
#pragma unroll
  for (int r = 0; r < 8; r++) {
    int o = obase + r;
    float y = acc[r] + bias[o];
    out[o * S_TOT + s] = (x[o * S_TOT + s] + y) * mk;
  }
}

__global__ void k_maskcopy(const float* __restrict__ mask, float* __restrict__ out) {
  int i = blockIdx.x * 256 + threadIdx.x;
  if (i < S_TOT) out[i] = mask[i];
}

extern "C" void kernel_launch(void* const* d_in, const int* in_sizes, int n_in,
                              void* d_out, int out_size, void* d_ws, size_t ws_size,
                              hipStream_t stream) {
  const float* x      = (const float*)d_in[0];
  const float* mask   = (const float*)d_in[1];
  const float* gn_w   = (const float*)d_in[2];
  const float* gn_b   = (const float*)d_in[3];
  const float* qkv_w  = (const float*)d_in[4];
  const float* qkv_b  = (const float*)d_in[5];
  const float* proj_w = (const float*)d_in[6];
  const float* proj_b = (const float*)d_in[7];
  float* out = (float*)d_out;

  const size_t PLANE = (size_t)S_TOT * C_TOT;      // 786432 elements
  unsigned short* xnT   = (unsigned short*)d_ws;   // bf16 [S][C]
  unsigned short* Qb    = xnT + PLANE;             // bf16 [H][S][32]
  unsigned short* Kb    = Qb + PLANE;              // bf16 [H][S][32]
  unsigned short* Vt    = Kb + PLANE;              // bf16 [H][32][S]
  unsigned short* attnT = Vt + PLANE;              // bf16 [S][C]
  unsigned short* Wq    = attnT + PLANE;           // bf16 [384][128]
  unsigned short* Wp    = Wq + 384 * C_TOT;        // bf16 [128][128]

  k_wprep<<<192, 256, 0, stream>>>(qkv_w, 384 * C_TOT, proj_w, C_TOT * C_TOT, Wq, Wp);
  k_groupnorm<<<8, 256, 0, stream>>>(x, gn_w, gn_b, xnT);
  k_qkv<<<2304, 128, 0, stream>>>(Wq, qkv_b, xnT, Qb, Kb, Vt);
  k_attn<<<dim3(96, 4), 128, 0, stream>>>(Qb, Kb, Vt, attnT);
  k_proj<<<768, 128, 0, stream>>>(Wp, proj_b, attnT, x, mask, out);
  k_maskcopy<<<24, 256, 0, stream>>>(mask, out + PLANE);
}
